// LSTM_82291573391471
// MI455X (gfx1250) — matrix operand order repaired and checked by reading, not compiled, not run on hardware
//
#include <hip/hip_runtime.h>
#include <hip/hip_bf16.h>

// ---------------------------------------------------------------------------
// Fused 2-layer LSTM (H=32) + FC head for gfx1250 (MI455X), wave32 WMMA.
//   - one wave per block, each wave owns a 16-sample batch tile
//   - gates[16x128] computed with v_wmma_f32_16x16x32_bf16 (bf16 in, f32 acc)
//   - ALL weight fragments register-resident (32 frags = 256 VGPRs; gfx1250
//     waves address up to 1024 VGPRs) -> no LDS weight reloads in the loop
//   - c-state kept in registers (WMMA C/D layout), h shuffled via 2KB LDS
//   - both layers fused per timestep; no global intermediate
// ---------------------------------------------------------------------------

typedef __attribute__((ext_vector_type(16))) __bf16 v16bf;
typedef __attribute__((ext_vector_type(8)))  float  v8f;

#define LDS_FENCE() asm volatile("s_wait_dscnt 0" ::: "memory")

union Frag16 {
    v16bf  v;
    __bf16 e[16];
    uint4  q[2];
};

static constexpr int T_STEPS = 512;
static constexpr int IN_DIM  = 5;
static constexpr int H_DIM   = 32;
static constexpr int OUT_DIM = 2;

__device__ __forceinline__ float fsigmoid(float x) {
    x = fminf(fmaxf(x, -30.f), 30.f);
    return __builtin_amdgcn_rcpf(1.f + __expf(-x));
}
__device__ __forceinline__ float ftanh(float x) {
    x = fminf(fmaxf(x, -15.f), 15.f);
    float e = __expf(2.f * x);
    return (e - 1.f) * __builtin_amdgcn_rcpf(e + 1.f);
}

// Build the A-operand (16x32 bf16, M=batch row, K=hidden) from an LDS h tile.
// 16-bit A layout: lanes 0-15 hold row=lane, K {0..7,16..23};
// lanes 16-31 hold row=lane-16, K {8..15,24..31}. Both runs are contiguous.
__device__ __forceinline__ Frag16 loadAFrag(const __bf16 (*hb)[H_DIM], int lane) {
    const int row = lane & 15;
    const int kb  = (lane >> 4) ? 8 : 0;
    Frag16 f;
    f.q[0] = *(const uint4*)&hb[row][kb];
    f.q[1] = *(const uint4*)&hb[row][kb + 16];
    return f;
}

// Gather W[4H x Kact] (row-major, leading dim = ld) straight from global into
// register-resident B-operand fragments (K x 16): lanes 0-15 -> col n=lane,
// k {0..7,16..23}; lanes 16-31 -> col n=lane-16, k {8..15,24..31}. Zero-pad k >= Kact.
__device__ __forceinline__ void buildBFragsReg(const float* __restrict__ W, int ld, int Kact,
                                               Frag16 dst[8], int lane) {
    const int n     = lane & 15;
    const int kbase = (lane < 16) ? 0 : 8;
#pragma unroll
    for (int j = 0; j < 8; ++j) {
        const int g = j * 16 + n;
#pragma unroll
        for (int i = 0; i < 16; ++i) {
            const int k = kbase + (i & 7) + ((i >> 3) << 4);
            float v = 0.f;
            if (k < Kact) v = W[g * ld + k];
            dst[j].e[i] = (__bf16)v;
        }
    }
}

// One LSTM cell step for a 16-sample tile.
//   a_in : A fragment of the input contribution (x or h0), K padded to 32
//   a_h  : A fragment of previous hidden state
//   wih/whh : register-resident B fragment arrays [8]
//   bias : per-lane bias (b_ih+b_hh) for gate column n = lane%16, 8 tiles
//   cfr  : cell state, C/D layout, cfr[pass] covers hidden cols pass*16+n
//   hb   : LDS tile receiving new h (bf16), [16][32]
__device__ __forceinline__ void lstm_step(const Frag16& a_in, const Frag16& a_h,
                                          const Frag16 wih[8], const Frag16 whh[8],
                                          const float* bias, v8f cfr[2],
                                          __bf16 (*hb)[H_DIM], int lane) {
    const int n     = lane & 15;
    const int mbase = (lane >> 4) ? 8 : 0;
#pragma unroll
    for (int pass = 0; pass < 2; ++pass) {
        v8f acc[4];
#pragma unroll
        for (int q = 0; q < 4; ++q) {           // q: gate i,f,g,o
            const int j = q * 2 + pass;         // gate-column tile 16j..16j+15
            v8f a;
#pragma unroll
            for (int r = 0; r < 8; ++r) a[r] = bias[j];
            a = __builtin_amdgcn_wmma_f32_16x16x32_bf16(false, a_in.v, false, wih[j].v,
                                                        (short)0, a, false, false);
            a = __builtin_amdgcn_wmma_f32_16x16x32_bf16(false, a_h.v, false, whh[j].v,
                                                        (short)0, a, false, false);
            acc[q] = a;
        }
#pragma unroll
        for (int r = 0; r < 8; ++r) {
            float ig = fsigmoid(acc[0][r]);
            float fg = fsigmoid(acc[1][r]);
            float gg = ftanh(acc[2][r]);
            float og = fsigmoid(acc[3][r]);
            float c  = fg * cfr[pass][r] + ig * gg;
            cfr[pass][r] = c;
            float h  = og * ftanh(c);
            hb[mbase + r][pass * 16 + n] = (__bf16)h;
        }
    }
}

__global__ __launch_bounds__(32) void lstm2_fused_kernel(
    const float* __restrict__ x,
    const float* __restrict__ Wih0, const float* __restrict__ Whh0,
    const float* __restrict__ bih0, const float* __restrict__ bhh0,
    const float* __restrict__ Wih1, const float* __restrict__ Whh1,
    const float* __restrict__ bih1, const float* __restrict__ bhh1,
    const float* __restrict__ Wfc,  const float* __restrict__ bfc,
    float* __restrict__ out) {
    // LDS: only the two h shuffle tiles (D-layout -> A-layout transpose).
    __shared__ alignas(16) __bf16 hb0[16][H_DIM];
    __shared__ alignas(16) __bf16 hb1[16][H_DIM];

    const int lane      = threadIdx.x;
    const int tile_base = blockIdx.x * 16;            // first batch sample of tile
    const int n         = lane & 15;

    // ---- init: register-resident weight fragments, bias regs, zero states ----
    Frag16 wih0[8], whh0[8], wih1[8], whh1[8];
    buildBFragsReg(Wih0, IN_DIM, IN_DIM, wih0, lane);
    buildBFragsReg(Whh0, H_DIM,  H_DIM,  whh0, lane);
    buildBFragsReg(Wih1, H_DIM,  H_DIM,  wih1, lane);
    buildBFragsReg(Whh1, H_DIM,  H_DIM,  whh1, lane);

    float bias0[8], bias1[8];
#pragma unroll
    for (int j = 0; j < 8; ++j) {
        bias0[j] = bih0[j * 16 + n] + bhh0[j * 16 + n];
        bias1[j] = bih1[j * 16 + n] + bhh1[j * 16 + n];
    }

    // zero h tiles (each lane clears a strided set of elements)
    {
        __bf16 z = (__bf16)0.f;
        for (int i = lane; i < 16 * H_DIM; i += 32) {
            (&hb0[0][0])[i] = z;
            (&hb1[0][0])[i] = z;
        }
    }

    v8f c0[2], c1[2];
#pragma unroll
    for (int p = 0; p < 2; ++p)
#pragma unroll
        for (int r = 0; r < 8; ++r) { c0[p][r] = 0.f; c1[p][r] = 0.f; }

    LDS_FENCE();

    // x A-fragment: only K=0..4 valid (held by lanes 0-15); rest stays zero.
    Frag16 ax0;
#pragma unroll
    for (int i = 0; i < 16; ++i) ax0.e[i] = (__bf16)0.f;

    // ---- recurrence over T, both layers fused per step ----
    for (int t = 0; t < T_STEPS; ++t) {
        // layer 0 input: x[b][t][0..4]
        if (lane < 16) {
            const float* xp = x + ((size_t)(tile_base + lane) * T_STEPS + t) * IN_DIM;
#pragma unroll
            for (int k = 0; k < IN_DIM; ++k) ax0.e[k] = (__bf16)xp[k];
        }
        Frag16 ah0 = loadAFrag(hb0, lane);            // h0^{t-1}
        lstm_step(ax0, ah0, wih0, whh0, bias0, c0, hb0, lane);
        LDS_FENCE();

        // layer 1 consumes h0^t immediately
        Frag16 ax1 = loadAFrag(hb0, lane);            // h0^t
        Frag16 ah1 = loadAFrag(hb1, lane);            // h1^{t-1}
        lstm_step(ax1, ah1, wih1, whh1, bias1, c1, hb1, lane);
        LDS_FENCE();
    }

    // ---- FC head: out[b] = h1_last @ Wfc^T + bfc ----
    if (lane < 16) {
        const int b = tile_base + lane;
        float a0 = bfc[0], a1 = bfc[1];
#pragma unroll
        for (int h = 0; h < H_DIM; ++h) {
            float hv = (float)hb1[lane][h];
            a0 += hv * Wfc[h];                        // Wfc[0][h]
            a1 += hv * Wfc[H_DIM + h];                // Wfc[1][h]
        }
        out[(size_t)b * OUT_DIM + 0] = a0;
        out[(size_t)b * OUT_DIM + 1] = a1;
    }
}

extern "C" void kernel_launch(void* const* d_in, const int* in_sizes, int n_in,
                              void* d_out, int out_size, void* d_ws, size_t ws_size,
                              hipStream_t stream) {
    (void)in_sizes; (void)n_in; (void)out_size; (void)d_ws; (void)ws_size;
    const float* x    = (const float*)d_in[0];
    const float* Wih0 = (const float*)d_in[1];
    const float* Whh0 = (const float*)d_in[2];
    const float* bih0 = (const float*)d_in[3];
    const float* bhh0 = (const float*)d_in[4];
    const float* Wih1 = (const float*)d_in[5];
    const float* Whh1 = (const float*)d_in[6];
    const float* bih1 = (const float*)d_in[7];
    const float* bhh1 = (const float*)d_in[8];
    const float* Wfc  = (const float*)d_in[9];
    const float* bfc  = (const float*)d_in[10];
    float* out = (float*)d_out;

    // B=2048 samples, 16-sample tile per wave, one wave per block.
    dim3 grid(2048 / 16);
    dim3 block(32);
    lstm2_fused_kernel<<<grid, block, 0, stream>>>(
        x, Wih0, Whh0, bih0, bhh0, Wih1, Whh1, bih1, bhh1, Wfc, bfc, out);
}
